// VerticalSpeakerMemoryTransformer_57028575756934
// MI455X (gfx1250) — compile-verified, matrix operand order checked by reading, not run on hardware
//
#include <hip/hip_runtime.h>
#include <hip/hip_bf16.h>
#include <math.h>

// ---------------- model constants ----------------
#define Bq   8
#define Tq   64
#define Lq   64
#define Hq   768
#define Fq   3072
#define NLq  4
#define NHq  12
#define CHq  4
#define Sq   4
#define H3q  2304
#define BTq  (Bq*Tq)          // 512
#define BTLq (Bq*Tq*Lq)       // 32768
#define EPSq 1e-5f

// ---------------- types ----------------
typedef __attribute__((ext_vector_type(16))) __bf16 v16bf;
typedef __attribute__((ext_vector_type(8)))  __bf16 v8bf;
typedef __attribute__((ext_vector_type(8)))  float  v8f;
typedef int v4i_t __attribute__((vector_size(16)));   // <4 x i32>, matches builtin param

__device__ __forceinline__ unsigned short f2bf(float f) {
    unsigned int u = __float_as_uint(f);
    unsigned int r = (u + 0x7FFFu + ((u >> 16) & 1u)) >> 16;
    return (unsigned short)r;
}
__device__ __forceinline__ float bf2f(unsigned short h) {
    return __uint_as_float(((unsigned int)h) << 16);
}
__device__ __forceinline__ float gelu_exact(float x) {
    return 0.5f * x * (1.0f + erff(x * 0.70710678118654752440f));
}

// CDNA5 async global->LDS copy path (guarded; sync fallback otherwise)
#if defined(__gfx1250__) && __has_builtin(__builtin_amdgcn_global_load_async_to_lds_b128) && __has_builtin(__builtin_amdgcn_s_wait_asynccnt)
#define HAVE_ASYNC_LDS 1
#define ASYNC_CP16(gsrc, ldst)                                  \
    __builtin_amdgcn_global_load_async_to_lds_b128(             \
        (v4i_t*)(gsrc), (v4i_t*)(ldst), 0, 0)
#else
#define HAVE_ASYNC_LDS 0
#endif

// =====================================================================
// WMMA bf16 GEMM:  C[M,N] = A[M,K](bf16,row) * Bt[N,K](bf16,row) + bias
// (B operand supplied PRE-TRANSPOSED: row n holds column n of B, K-contig)
// block tile 128x128, K-step 32; 256 threads = 8 waves; wave tile 32x64
// =====================================================================
#define BM 128
#define BN 128
#define BK 32

__global__ __launch_bounds__(256) void gemm_bf16_wmma(
    const unsigned short* __restrict__ A, int lda,
    const unsigned short* __restrict__ Bt, int ldb,   // ldb = K-stride of Bt row
    const float* __restrict__ bias,
    float* __restrict__ outF, unsigned short* __restrict__ outB, int ldc,
    int Mdim, int Ndim, int Kdim, int act)
{
    __shared__ unsigned short As[BM][BK];   // [m][k]
    __shared__ unsigned short Bs[BN][BK];   // [n][k]

    const int tid  = threadIdx.x;
    const int lane = tid & 31;
    const int wave = tid >> 5;
    const int wm   = wave & 3;    // 4 waves along M
    const int wn   = wave >> 2;   // 2 waves along N
    const int blockM = blockIdx.y * BM;
    const int blockN = blockIdx.x * BN;

    v8f acc[2][4];
    const v8f zacc = {0.f,0.f,0.f,0.f,0.f,0.f,0.f,0.f};
#pragma unroll
    for (int i = 0; i < 2; ++i)
#pragma unroll
        for (int j = 0; j < 4; ++j) acc[i][j] = zacc;

    const int kb = (lane >> 4) * 8;   // K sub-base per ISA 16-bit operand layout
    const int rr = lane & 15;
    const uint4 zq = {0u,0u,0u,0u};

    for (int k0 = 0; k0 < Kdim; k0 += BK) {
        // ---- stage A tile: 128x32 halves = 512 uint4, 2 per thread ----
#pragma unroll
        for (int i = 0; i < 2; ++i) {
            int idx  = tid + i * 256;      // 0..511
            int row  = idx >> 2;
            int ch   = idx & 3;
            int grow = blockM + row;
            if (grow < Mdim) {
#if HAVE_ASYNC_LDS
                ASYNC_CP16(A + (size_t)grow * lda + k0 + ch * 8, &As[row][ch * 8]);
#else
                *(uint4*)(&As[row][ch * 8]) =
                    *(const uint4*)(A + (size_t)grow * lda + k0 + ch * 8);
#endif
            } else {
                *(uint4*)(&As[row][ch * 8]) = zq;
            }
        }
        // ---- stage B tile: identical pattern from pre-transposed weights ----
#pragma unroll
        for (int i = 0; i < 2; ++i) {
            int idx  = tid + i * 256;
            int row  = idx >> 2;
            int ch   = idx & 3;
            int gcol = blockN + row;
            if (gcol < Ndim) {
#if HAVE_ASYNC_LDS
                ASYNC_CP16(Bt + (size_t)gcol * ldb + k0 + ch * 8, &Bs[row][ch * 8]);
#else
                *(uint4*)(&Bs[row][ch * 8]) =
                    *(const uint4*)(Bt + (size_t)gcol * ldb + k0 + ch * 8);
#endif
            } else {
                *(uint4*)(&Bs[row][ch * 8]) = zq;
            }
        }
        // prefetch next A tile (global_prefetch_b8 path)
        if (k0 + BK < Kdim) {
            int grow = blockM + (tid >> 1);
            if (grow < Mdim)
                __builtin_prefetch(A + (size_t)grow * lda + k0 + BK, 0, 0);
        }
#if HAVE_ASYNC_LDS
        __builtin_amdgcn_s_wait_asynccnt(0);
#endif
        __syncthreads();

        // ---- load fragments (two b128 LDS reads per fragment) ----
        v16bf af[2], bf[4];
#pragma unroll
        for (int ti = 0; ti < 2; ++ti) {
            int row = wm * 32 + ti * 16 + rr;
            v8bf lo = *(const v8bf*)(&As[row][kb]);
            v8bf hi = *(const v8bf*)(&As[row][16 + kb]);
#pragma unroll
            for (int e = 0; e < 8; ++e) { af[ti][e] = lo[e]; af[ti][e + 8] = hi[e]; }
        }
#pragma unroll
        for (int tj = 0; tj < 4; ++tj) {
            int col = wn * 64 + tj * 16 + rr;
            v8bf lo = *(const v8bf*)(&Bs[col][kb]);
            v8bf hi = *(const v8bf*)(&Bs[col][16 + kb]);
#pragma unroll
            for (int e = 0; e < 8; ++e) { bf[tj][e] = lo[e]; bf[tj][e + 8] = hi[e]; }
        }
        // ---- 8 WMMAs per K-step ----
#pragma unroll
        for (int ti = 0; ti < 2; ++ti)
#pragma unroll
            for (int tj = 0; tj < 4; ++tj)
                acc[ti][tj] = __builtin_amdgcn_wmma_f32_16x16x32_bf16(
                    false, af[ti], false, bf[tj], (short)0, acc[ti][tj], false, false);
        __syncthreads();
    }

    // ---- epilogue: C/D layout lanes0-15 M=r, lanes16-31 M=8+r ----
    const int halfsel = lane >> 4;
    const int nloc    = lane & 15;
#pragma unroll
    for (int ti = 0; ti < 2; ++ti) {
#pragma unroll
        for (int tj = 0; tj < 4; ++tj) {
#pragma unroll
            for (int r = 0; r < 8; ++r) {
                int grow = blockM + wm * 32 + ti * 16 + r + halfsel * 8;
                int gcol = blockN + wn * 64 + tj * 16 + nloc;
                if (grow < Mdim && gcol < Ndim) {
                    float v = acc[ti][tj][r];
                    if (bias) v += bias[gcol];
                    if (act == 1) v = gelu_exact(v);
                    size_t o = (size_t)grow * ldc + gcol;
                    if (outF) outF[o] = v;
                    if (outB) outB[o] = f2bf(v);
                }
            }
        }
    }
}

// =====================================================================
// elementwise / reduction kernels
// =====================================================================
// cast + transpose: src [NL][K][N] f32 -> dst [NL][N][K] bf16
__global__ __launch_bounds__(256) void castT_kernel(
    const float* __restrict__ src, unsigned short* __restrict__ dst, int K, int N)
{
    int kl = blockIdx.x;          // l*K + k
    int l = kl / K, k = kl - l * K;
    const float* s = src + ((size_t)l * K + k) * N;
    unsigned short* d = dst + (size_t)l * N * K + k;
    for (int n = threadIdx.x; n < N; n += 256)
        d[(size_t)n * K] = f2bf(s[n]);
}

__global__ __launch_bounds__(256) void embed_kernel(
    const int* __restrict__ ids, const float* __restrict__ tok,
    const float* __restrict__ pos, const float* __restrict__ turn,
    float* __restrict__ X)
{
    int btl = blockIdx.x;
    int id  = ids[btl];
    int l   = btl & 63;
    int t   = (btl >> 6) & 63;
#pragma unroll
    for (int i = 0; i < 3; ++i) {
        int h = threadIdx.x + i * 256;
        X[(size_t)btl * Hq + h] =
            tok[(size_t)id * Hq + h] + pos[(size_t)l * Hq + h] + turn[(size_t)t * Hq + h];
    }
}

__global__ __launch_bounds__(256) void init_M_kernel(float* Mf, unsigned short* Mb)
{
    int i = blockIdx.x * 256 + threadIdx.x;
    if (i < Bq * Sq * Hq) { Mf[i] = 0.f; Mb[i] = 0; }
}

__global__ void counts_kernel(const int* __restrict__ spk, float* __restrict__ counts)
{
    int i = threadIdx.x;
    if (i < Bq * Sq) {
        int b = i >> 2, s = i & 3, c = 0;
        for (int t = 0; t < Tq; ++t) c += (spk[b * Tq + t] == s);
        counts[i] = (float)c;
    }
}

__global__ __launch_bounds__(256) void mt_kernel(
    const int* __restrict__ spk, const float* __restrict__ Mf, float* __restrict__ mt)
{
    int bt = blockIdx.x;
    int b  = bt >> 6;
    int s  = spk[bt];
#pragma unroll
    for (int i = 0; i < 3; ++i) {
        int h = threadIdx.x + i * 256;
        mt[(size_t)bt * Hq + h] = Mf[((size_t)(b * Sq + s)) * Hq + h];
    }
}

__global__ __launch_bounds__(256) void hb_kernel(
    const float* __restrict__ X, const float* __restrict__ mt,
    const int* __restrict__ amask, unsigned short* __restrict__ hb)
{
    int btl = blockIdx.x;
    int bt  = btl >> 6;
    float mk = (float)amask[btl];
#pragma unroll
    for (int i = 0; i < 3; ++i) {
        int h = threadIdx.x + i * 256;
        float v = (X[(size_t)btl * Hq + h] + mt[(size_t)bt * Hq + h]) * mk;
        hb[(size_t)btl * Hq + h] = f2bf(v);
    }
}

// per-(b,t,head) self-attention over L=64 tokens, dh=64
__global__ __launch_bounds__(64) void self_attn_kernel(
    const unsigned short* __restrict__ qkv, unsigned short* __restrict__ ao)
{
    int head = blockIdx.x % NHq;
    int bt   = blockIdx.x / NHq;
    size_t base = (size_t)bt * Lq * H3q;

    __shared__ float Ks[64][65];
    __shared__ float Vs[64][65];
    int q = threadIdx.x;   // 0..63 = query row / staging row
#pragma unroll 8
    for (int d = 0; d < 64; ++d) {
        Ks[q][d] = bf2f(qkv[base + (size_t)q * H3q + Hq     + head * 64 + d]);
        Vs[q][d] = bf2f(qkv[base + (size_t)q * H3q + 2 * Hq + head * 64 + d]);
    }
    __syncthreads();

    float qv[64];
#pragma unroll
    for (int d = 0; d < 64; ++d)
        qv[d] = bf2f(qkv[base + (size_t)q * H3q + head * 64 + d]);

    float s[64];
    float mx = -3.4e38f;
    for (int kk = 0; kk < 64; ++kk) {
        float dot = 0.f;
#pragma unroll
        for (int d = 0; d < 64; ++d) dot += qv[d] * Ks[kk][d];
        dot *= 0.125f;               // 1/sqrt(64)
        s[kk] = dot;
        mx = fmaxf(mx, dot);
    }
    float sum = 0.f;
    for (int kk = 0; kk < 64; ++kk) { s[kk] = __expf(s[kk] - mx); sum += s[kk]; }
    float inv = 1.f / sum;

    size_t orow = ((size_t)bt * Lq + q) * Hq + head * 64;
    for (int d = 0; d < 64; ++d) {
        float o = 0.f;
#pragma unroll
        for (int kk = 0; kk < 64; ++kk) o += s[kk] * Vs[kk][d];
        ao[orow + d] = f2bf(o * inv);
    }
}

// LayerNorm over H=768 of (a + res), optional f32 and bf16 outputs
__global__ __launch_bounds__(256) void ln_kernel(
    const float* __restrict__ a, const float* __restrict__ res,
    const float* __restrict__ g, const float* __restrict__ bb,
    float* __restrict__ outF, unsigned short* __restrict__ outB)
{
    int row = blockIdx.x, tid = threadIdx.x;
    const float* ap = a + (size_t)row * Hq;
    const float* rp = res ? res + (size_t)row * Hq : nullptr;
    __shared__ float red[256];

    float local[3];
    float sum = 0.f;
#pragma unroll
    for (int i = 0; i < 3; ++i) {
        int h = tid + i * 256;
        float x = ap[h] + (rp ? rp[h] : 0.f);
        local[i] = x; sum += x;
    }
    red[tid] = sum; __syncthreads();
    for (int off = 128; off > 0; off >>= 1) {
        if (tid < off) red[tid] += red[tid + off];
        __syncthreads();
    }
    float mean = red[0] * (1.f / Hq);
    __syncthreads();

    float vs = 0.f;
#pragma unroll
    for (int i = 0; i < 3; ++i) { float d = local[i] - mean; vs += d * d; }
    red[tid] = vs; __syncthreads();
    for (int off = 128; off > 0; off >>= 1) {
        if (tid < off) red[tid] += red[tid + off];
        __syncthreads();
    }
    float rstd = rsqrtf(red[0] * (1.f / Hq) + EPSq);

#pragma unroll
    for (int i = 0; i < 3; ++i) {
        int h = tid + i * 256;
        float y = (local[i] - mean) * rstd * g[h] + bb[h];
        size_t o = (size_t)row * Hq + h;
        if (outF) outF[o] = y;
        if (outB) outB[o] = f2bf(y);
    }
}

__global__ __launch_bounds__(256) void tc_kernel(
    const float* __restrict__ X, float* __restrict__ tc, unsigned short* __restrict__ tcb)
{
    int bt = blockIdx.x;
#pragma unroll
    for (int i = 0; i < 3; ++i) {
        int h = threadIdx.x + i * 256;
        float sum = 0.f;
        for (int l = 0; l < Lq; ++l)
            sum += X[((size_t)bt * Lq + l) * Hq + h];
        float v = sum * (1.f / Lq);
        tc[(size_t)bt * Hq + h]  = v;
        tcb[(size_t)bt * Hq + h] = f2bf(v);
    }
}

// cross-attn: q[512,768] vs kv[32,1536] (k|v), CH=4 heads, dh=192
__global__ __launch_bounds__(256) void cross_attn_kernel(
    const float* __restrict__ q, const float* __restrict__ kv, unsigned short* __restrict__ ccb)
{
    int bt = blockIdx.x;
    int b  = bt >> 6;
    int tid = threadIdx.x;
    __shared__ float sc[16];   // [head*4 + s]

    if (tid < 16) {
        int h = tid >> 2, s = tid & 3;
        const float* qp = q  + (size_t)bt * Hq + h * 192;
        const float* kp = kv + ((size_t)(b * Sq + s)) * (2 * Hq) + h * 192;
        float d = 0.f;
        for (int i = 0; i < 192; ++i) d += qp[i] * kp[i];
        sc[tid] = d * 0.07216878364870323f;   // 1/sqrt(192)
    }
    __syncthreads();
    if (tid < 4) {
        float a0 = sc[tid*4], a1 = sc[tid*4+1], a2 = sc[tid*4+2], a3 = sc[tid*4+3];
        float m = fmaxf(fmaxf(a0, a1), fmaxf(a2, a3));
        float e0 = __expf(a0-m), e1 = __expf(a1-m), e2 = __expf(a2-m), e3 = __expf(a3-m);
        float inv = 1.f / (e0+e1+e2+e3);
        sc[tid*4] = e0*inv; sc[tid*4+1] = e1*inv; sc[tid*4+2] = e2*inv; sc[tid*4+3] = e3*inv;
    }
    __syncthreads();
#pragma unroll
    for (int i = 0; i < 3; ++i) {
        int idx = tid + i * 256;  // 0..767
        int h = idx / 192, d = idx % 192;
        float o = 0.f;
#pragma unroll
        for (int s = 0; s < 4; ++s)
            o += sc[h*4 + s] * kv[((size_t)(b * Sq + s)) * (2 * Hq) + Hq + h * 192 + d];
        ccb[(size_t)bt * Hq + idx] = f2bf(o);
    }
}

__global__ __launch_bounds__(256) void concat_kernel(
    const float* __restrict__ tc, const float* __restrict__ mt,
    const float* __restrict__ cco, unsigned short* __restrict__ fused)
{
    int bt = blockIdx.x;
#pragma unroll
    for (int i = 0; i < 9; ++i) {
        int j = threadIdx.x + i * 256;   // 0..2303
        float v;
        if (j < Hq)            v = tc [(size_t)bt * Hq + j];
        else if (j < 2 * Hq)   v = mt [(size_t)bt * Hq + (j - Hq)];
        else                   v = cco[(size_t)bt * Hq + (j - 2 * Hq)];
        fused[(size_t)bt * H3q + j] = f2bf(v);
    }
}

__global__ __launch_bounds__(256) void scatter_kernel(
    const int* __restrict__ spk, const float* __restrict__ upd,
    const float* __restrict__ counts, float* __restrict__ Mf, unsigned short* __restrict__ Mb)
{
    int bs = blockIdx.x;
    int b = bs >> 2, s = bs & 3;
    float c = counts[bs];
#pragma unroll
    for (int i = 0; i < 3; ++i) {
        int h = threadIdx.x + i * 256;
        float sum = 0.f;
        for (int t = 0; t < Tq; ++t)
            if (spk[b * Tq + t] == s)
                sum += upd[((size_t)(b * Tq + t)) * Hq + h];
        size_t off = (size_t)bs * Hq + h;
        float m = Mf[off];
        if (c > 0.f) m = sum / c;
        Mf[off] = m;
        Mb[off] = f2bf(m);
    }
}

// =====================================================================
// host launcher
// =====================================================================
static inline void gemm(hipStream_t st, const unsigned short* A, int lda,
                        const unsigned short* Bt, int ldb, const float* bias,
                        float* outF, unsigned short* outB, int ldc,
                        int M, int N, int K, int act)
{
    dim3 g((N + BN - 1) / BN, (M + BM - 1) / BM);
    gemm_bf16_wmma<<<g, dim3(256), 0, st>>>(A, lda, Bt, ldb, bias, outF, outB, ldc, M, N, K, act);
}

extern "C" void kernel_launch(void* const* d_in, const int* in_sizes, int n_in,
                              void* d_out, int out_size, void* d_ws, size_t ws_size,
                              hipStream_t stream)
{
    // ---- inputs ----
    const int*   ids    = (const int*)  d_in[0];
    const int*   amask  = (const int*)  d_in[1];
    const int*   spk    = (const int*)  d_in[2];
    const float* tok    = (const float*)d_in[3];
    const float* pos    = (const float*)d_in[4];
    const float* turn   = (const float*)d_in[5];
    const float* mha_in_w  = (const float*)d_in[6];
    const float* mha_in_b  = (const float*)d_in[7];
    const float* mha_out_w = (const float*)d_in[8];
    const float* mha_out_b = (const float*)d_in[9];
    const float* ffn_w1 = (const float*)d_in[10];
    const float* ffn_b1 = (const float*)d_in[11];
    const float* ffn_w2 = (const float*)d_in[12];
    const float* ffn_b2 = (const float*)d_in[13];
    const float* ln1_g  = (const float*)d_in[14];
    const float* ln1_b  = (const float*)d_in[15];
    const float* ln2_g  = (const float*)d_in[16];
    const float* ln2_b  = (const float*)d_in[17];
    const float* gate_w = (const float*)d_in[18];
    const float* gate_b = (const float*)d_in[19];
    const float* sn_g   = (const float*)d_in[20];
    const float* sn_b   = (const float*)d_in[21];
    const float* ca_in_w  = (const float*)d_in[22];
    const float* ca_in_b  = (const float*)d_in[23];
    const float* ca_out_w = (const float*)d_in[24];
    const float* ca_out_b = (const float*)d_in[25];

    // ---- outputs live in d_out: X then M ----
    float* X  = (float*)d_out;
    float* Mf = X + (size_t)BTLq * Hq;

    // ---- workspace carve ----
    char* wsp = (char*)d_ws;
    size_t off = 0;
    auto alloc = [&](size_t bytes) -> void* {
        void* p = wsp + off;
        off += (bytes + 255) & ~(size_t)255;
        return p;
    };
    unsigned short* hb    = (unsigned short*)alloc((size_t)BTLq * Hq  * 2);
    unsigned short* qkvb  = (unsigned short*)alloc((size_t)BTLq * H3q * 2);
    unsigned short* aob   = (unsigned short*)alloc((size_t)BTLq * Hq  * 2);
    float*          proj  = (float*)         alloc((size_t)BTLq * Hq  * 4);
    float*          H1    = (float*)         alloc((size_t)BTLq * Hq  * 4);
    unsigned short* h_b   = (unsigned short*)alloc((size_t)BTLq * Hq  * 2);
    unsigned short* g_b   = (unsigned short*)alloc((size_t)BTLq * Fq  * 2);
    float*          ffn2  = (float*)         alloc((size_t)BTLq * Hq  * 4);
    unsigned short* wInB  = (unsigned short*)alloc((size_t)NLq * Hq * H3q * 2);
    unsigned short* wOutB = (unsigned short*)alloc((size_t)NLq * Hq * Hq  * 2);
    unsigned short* w1B   = (unsigned short*)alloc((size_t)NLq * Hq * Fq  * 2);
    unsigned short* w2B   = (unsigned short*)alloc((size_t)NLq * Fq * Hq  * 2);
    unsigned short* gateB = (unsigned short*)alloc((size_t)NLq * H3q * Hq * 2);
    unsigned short* caInB = (unsigned short*)alloc((size_t)NLq * Hq * H3q * 2);
    unsigned short* caOutB= (unsigned short*)alloc((size_t)NLq * Hq * Hq  * 2);
    float*          tc    = (float*)         alloc((size_t)BTq * Hq * 4);
    unsigned short* tcb   = (unsigned short*)alloc((size_t)BTq * Hq * 2);
    float*          mt    = (float*)         alloc((size_t)BTq * Hq * 4);
    float*          qca   = (float*)         alloc((size_t)BTq * Hq * 4);
    float*          kvca  = (float*)         alloc((size_t)Bq * Sq * 2 * Hq * 4);
    unsigned short* ccb   = (unsigned short*)alloc((size_t)BTq * Hq * 2);
    float*          cco   = (float*)         alloc((size_t)BTq * Hq * 4);
    unsigned short* fusedb= (unsigned short*)alloc((size_t)BTq * H3q * 2);
    float*          gatef = (float*)         alloc((size_t)BTq * Hq * 4);
    float*          updf  = (float*)         alloc((size_t)BTq * Hq * 4);
    unsigned short* Mb    = (unsigned short*)alloc((size_t)Bq * Sq * Hq * 2);
    float*          countsf=(float*)         alloc((size_t)Bq * Sq * 4);
    (void)ws_size; (void)n_in; (void)in_sizes; (void)out_size;

    // ---- weight cast+transpose (once per call): [K][N] f32 -> [N][K] bf16 ----
    auto castT = [&](const float* s, unsigned short* d, int K, int N) {
        castT_kernel<<<NLq * K, 256, 0, stream>>>(s, d, K, N);
    };
    castT(mha_in_w,  wInB,  Hq,  H3q);   // -> [2304][768]
    castT(mha_out_w, wOutB, Hq,  Hq);    // -> [768][768]
    castT(ffn_w1,    w1B,   Hq,  Fq);    // -> [3072][768]
    castT(ffn_w2,    w2B,   Fq,  Hq);    // -> [768][3072]
    castT(gate_w,    gateB, H3q, Hq);    // -> [768][2304]
    castT(ca_in_w,   caInB, Hq,  H3q);   // -> [2304][768]
    castT(ca_out_w,  caOutB,Hq,  Hq);    // -> [768][768]

    // ---- setup ----
    init_M_kernel<<<(Bq*Sq*Hq + 255)/256, 256, 0, stream>>>(Mf, Mb);
    counts_kernel<<<1, 32, 0, stream>>>(spk, countsf);
    embed_kernel<<<BTLq, 256, 0, stream>>>(ids, tok, pos, turn, X);

    // ---- layers ----
    for (int l = 0; l < NLq; ++l) {
        // memory gather & masked input
        mt_kernel<<<BTq, 256, 0, stream>>>(spk, Mf, mt);
        hb_kernel<<<BTLq, 256, 0, stream>>>(X, mt, amask, hb);

        // self-attention
        gemm(stream, hb, Hq, wInB + (size_t)l * Hq * H3q, Hq,
             mha_in_b + (size_t)l * H3q, nullptr, qkvb, H3q, BTLq, H3q, Hq, 0);
        self_attn_kernel<<<BTq * NHq, 64, 0, stream>>>(qkvb, aob);
        gemm(stream, aob, Hq, wOutB + (size_t)l * Hq * Hq, Hq,
             mha_out_b + (size_t)l * Hq, proj, nullptr, Hq, BTLq, Hq, Hq, 0);
        ln_kernel<<<BTLq, 256, 0, stream>>>(proj, X, ln1_g + (size_t)l * Hq,
                                            ln1_b + (size_t)l * Hq, H1, h_b);

        // FFN
        gemm(stream, h_b, Hq, w1B + (size_t)l * Hq * Fq, Hq,
             ffn_b1 + (size_t)l * Fq, nullptr, g_b, Fq, BTLq, Fq, Hq, 1);
        gemm(stream, g_b, Fq, w2B + (size_t)l * Fq * Hq, Fq,
             ffn_b2 + (size_t)l * Hq, ffn2, nullptr, Hq, BTLq, Hq, Fq, 0);
        ln_kernel<<<BTLq, 256, 0, stream>>>(ffn2, H1, ln2_g + (size_t)l * Hq,
                                            ln2_b + (size_t)l * Hq, X, nullptr);

        // speaker-memory update path
        tc_kernel<<<BTq, 256, 0, stream>>>(X, tc, tcb);
        gemm(stream, tcb, Hq, caInB + (size_t)l * Hq * H3q, Hq,
             ca_in_b + (size_t)l * H3q, qca, nullptr, Hq, BTq, Hq, Hq, 0);
        gemm(stream, Mb, Hq, caInB + (size_t)l * Hq * H3q + (size_t)Hq * Hq, Hq,
             ca_in_b + (size_t)l * H3q + Hq, kvca, nullptr, 2 * Hq,
             Bq * Sq, 2 * Hq, Hq, 0);
        cross_attn_kernel<<<BTq, 256, 0, stream>>>(qca, kvca, ccb);
        gemm(stream, ccb, Hq, caOutB + (size_t)l * Hq * Hq, Hq,
             ca_out_b + (size_t)l * Hq, cco, nullptr, Hq, BTq, Hq, Hq, 0);
        concat_kernel<<<BTq, 256, 0, stream>>>(tc, mt, cco, fusedb);
        gemm(stream, fusedb, H3q, gateB + (size_t)l * Hq * H3q, H3q,
             gate_b + (size_t)l * Hq, gatef, nullptr, Hq, BTq, Hq, H3q, 0);
        ln_kernel<<<BTq, 256, 0, stream>>>(gatef, mt, sn_g + (size_t)l * Hq,
                                           sn_b + (size_t)l * Hq, updf, nullptr);
        scatter_kernel<<<Bq * Sq, 256, 0, stream>>>(spk, updf, countsf, Mf, Mb);
    }
}
